// SimpleLSTMCell_45887430590719
// MI455X (gfx1250) — compile-verified, matrix operand order checked by reading, not run on hardware
//
#include <hip/hip_runtime.h>
#include <hip/hip_bf16.h>

// Problem constants (match reference)
#define HID   1024
#define BATCH 1024
#define TPAST 96
#define FUT   32
#define TTOT  128

// Use CDNA5 async global->LDS copies for tile staging (ASYNCcnt path).
// Set to 0 to fall back to global_load + ds_store staging.
#define USE_ASYNC_LDS 1

typedef __attribute__((ext_vector_type(16))) __bf16 v16bf;
typedef __attribute__((ext_vector_type(8)))  float  v8f;

union Frag { uint4 q[2]; v16bf v; };

__device__ __forceinline__ unsigned short f2bf(float f) {
    unsigned u = __float_as_uint(f);
    u += 0x7FFFu + ((u >> 16) & 1u);        // round-to-nearest-even
    return (unsigned short)(u >> 16);
}
__device__ __forceinline__ float bf2f(unsigned short s) {
    return __uint_as_float(((unsigned)s) << 16);
}
__device__ __forceinline__ float sigm(float x) { return 1.0f / (1.0f + __expf(-x)); }

#if USE_ASYNC_LDS
// 16B async copies: LDS[lds_byte + imm] = MEM[saddr + vaddr_byte + imm]
// (VGLOBAL async-to-LDS, GVS mode; IOFFSET applies to both addresses.)
// NOTE: no "memory" clobber -- ordering is enforced by the volatile chain and
// by wait_async0()'s memory clobber + __syncthreads(); this keeps the
// scheduler free to pipeline WMMAs around the issue sites.
__device__ __forceinline__ void async_cp32(unsigned lds_byte, unsigned vaddr_byte,
                                           unsigned long long saddr) {
    asm volatile("global_load_async_to_lds_b128 %0, %1, %2 offset:0"
                 :: "v"(lds_byte), "v"(vaddr_byte), "s"(saddr));
    asm volatile("global_load_async_to_lds_b128 %0, %1, %2 offset:16"
                 :: "v"(lds_byte), "v"(vaddr_byte), "s"(saddr));
}
__device__ __forceinline__ void wait_async0() {
    asm volatile("s_wait_asynccnt 0x0" ::: "memory");
}
#endif

// 8 WMMAs for one 32-K chunk: 2 A row-fragments x 4 gates, in-place acc.
__device__ __forceinline__ void compute_chunk(const unsigned short* __restrict__ sHc,
                                              const unsigned short* __restrict__ sWc,
                                              int aRow0, int aRow1, int bBase,
                                              v8f acc[4][2], int half)
{
    Frag a0, a1;
    a0.q[0] = *(const uint4*)&sHc[aRow0 + half * 8];
    a0.q[1] = *(const uint4*)&sHc[aRow0 + 16 + half * 8];
    a1.q[0] = *(const uint4*)&sHc[aRow1 + half * 8];
    a1.q[1] = *(const uint4*)&sHc[aRow1 + 16 + half * 8];

#pragma unroll
    for (int g = 0; g < 4; ++g) {
        Frag b;
        const int brow = bBase + g * 32 * 40;
        b.q[0] = *(const uint4*)&sWc[brow + half * 8];
        b.q[1] = *(const uint4*)&sWc[brow + 16 + half * 8];
        acc[g][0] = __builtin_amdgcn_wmma_f32_16x16x32_bf16(
            false, a0.v, false, b.v, (short)0, acc[g][0], false, false);
        acc[g][1] = __builtin_amdgcn_wmma_f32_16x16x32_bf16(
            false, a1.v, false, b.v, (short)0, acc[g][1], false, false);
    }
}

// ---------------------------------------------------------------------------
// One LSTM step: gates = h @ W_hh^T (bf16 WMMA, f32 acc) + x*W_ih + bias,
// then the elementwise cell update, fused.
//
// Block: 256 thr = 8 waves. Tile: 128 batch-rows x 32 hidden-cols.
// Wave (waveM 0..3, waveN 0..1): 32 rows x 16 cols x 4 gates = 8 accumulators.
// h tile and W tile (4 gates x 32 cols x 32 K) staged through double-buffered
// LDS via async global->LDS copies; one barrier per K-chunk; K-loop manually
// unrolled x2 so each buffer reference is compile-time constant.
// ---------------------------------------------------------------------------
__global__ __launch_bounds__(256)
void lstm_step(const unsigned short* __restrict__ hprev,  // [B,H] bf16
               float* __restrict__ cstate,                // [B,H] f32 (in/out)
               unsigned short* __restrict__ hnext,        // [B,H] bf16
               const unsigned short* __restrict__ Wb,     // [4H,H] bf16 (row n = gate col)
               const float* __restrict__ Wih,             // [4H] (input dim == 1)
               const float* __restrict__ bias,            // [4H] (b_ih + b_hh)
               const float* __restrict__ xbase, int xstride)
{
    // Double-buffered tiles, padded row stride 40 shorts (80B) for bank spread
    __shared__ unsigned short sH[2][128 * 40];   // h rows [128 x 32K]
    __shared__ unsigned short sW[2][128 * 40];   // W rows: idx = gate*32 + col

    const int tid   = threadIdx.x;
    const int lane  = tid & 31;
    const int w     = tid >> 5;
    const int waveM = w >> 1;                // 0..3 -> 32-row slice
    const int waveN = w & 1;                 // 0..1 -> 16-col slice
    const int half  = lane >> 4;             // 0/1
    const int l16   = lane & 15;

    const int rowBase = blockIdx.y * 128;
    const int colBlk  = blockIdx.x * 32;
    const int colBase = colBlk + waveN * 16;

    v8f acc[4][2] = {};                      // [gate][row-fragment]

    // --- cooperative staging: thread t owns row = t>>1, 16-short half = t&1
    // (32 bytes per array per thread per K-chunk -> 2 async b128 each)
    const int srow  = tid >> 1;              // 0..127
    const int shalf = (tid & 1) * 16;        // shorts
    const int sOff  = srow * 40 + shalf;
    const int wgate = srow >> 5;             // 0..3
    const int wcol  = srow & 31;

#if USE_ASYNC_LDS
    // Per-thread byte offsets against the uniform SGPR base pointers
    const unsigned vaH = (unsigned)(((rowBase + srow) * HID + shalf) * 2);
    const unsigned vaW = (unsigned)(((wgate * HID + colBlk + wcol) * HID + shalf) * 2);
    const unsigned long long hbase = (unsigned long long)hprev;
    const unsigned long long wbase = (unsigned long long)Wb;
    // LDS byte addresses (generic LDS pointer keeps wave-relative offset in [31:0])
    const unsigned ldsH0 = (unsigned)(uintptr_t)&sH[0][sOff];
    const unsigned ldsH1 = (unsigned)(uintptr_t)&sH[1][sOff];
    const unsigned ldsW0 = (unsigned)(uintptr_t)&sW[0][sOff];
    const unsigned ldsW1 = (unsigned)(uintptr_t)&sW[1][sOff];

    // prologue: chunk 0 -> buffer 0
    async_cp32(ldsH0, vaH, hbase);
    async_cp32(ldsW0, vaW, wbase);
    wait_async0();
    __syncthreads();
#else
    const unsigned short* gH = hprev + (size_t)(rowBase + srow) * HID + shalf;
    const unsigned short* gW = Wb + (size_t)(wgate * HID + colBlk + wcol) * HID + shalf;
    *(uint4*)&sH[0][sOff + 0] = *(const uint4*)(gH + 0);
    *(uint4*)&sH[0][sOff + 8] = *(const uint4*)(gH + 8);
    *(uint4*)&sW[0][sOff + 0] = *(const uint4*)(gW + 0);
    *(uint4*)&sW[0][sOff + 8] = *(const uint4*)(gW + 8);
    __syncthreads();
#endif

    const int aRow0 = (waveM * 32 + l16) * 40;        // rows 0..15 of slice
    const int aRow1 = (waveM * 32 + 16 + l16) * 40;   // rows 16..31 of slice
    const int bBase = (waveN * 16 + l16) * 40;

#if USE_ASYNC_LDS
    for (int k = 0; k < 32; k += 2) {
        // ---- even chunk k: compute from buf0, prefetch k+1 -> buf1
        async_cp32(ldsH1, vaH + (unsigned)(k + 1) * 64u, hbase);
        async_cp32(ldsW1, vaW + (unsigned)(k + 1) * 64u, wbase);
        compute_chunk(sH[0], sW[0], aRow0, aRow1, bBase, acc, half);
        wait_async0();          // my buf1 writes landed
        __syncthreads();        // everyone's buf1 writes visible; buf0 reads done

        // ---- odd chunk k+1: compute from buf1, prefetch k+2 -> buf0
        if (k + 2 < 32) {
            async_cp32(ldsH0, vaH + (unsigned)(k + 2) * 64u, hbase);
            async_cp32(ldsW0, vaW + (unsigned)(k + 2) * 64u, wbase);
        }
        compute_chunk(sH[1], sW[1], aRow0, aRow1, bBase, acc, half);
        if (k + 2 < 32) {
            wait_async0();
            __syncthreads();
        }
    }
#else
    for (int k = 0; k < 32; ++k) {
        const int cur = k & 1;
        const int nxt = cur ^ 1;
        uint4 h0, h1, w0, w1;
        if (k < 31) {
            const int ko = (k + 1) * 32;
            h0 = *(const uint4*)(gH + ko);
            h1 = *(const uint4*)(gH + ko + 8);
            w0 = *(const uint4*)(gW + ko);
            w1 = *(const uint4*)(gW + ko + 8);
        }
        compute_chunk(sH[cur], sW[cur], aRow0, aRow1, bBase, acc, half);
        if (k < 31) {
            __syncthreads();
            *(uint4*)&sH[nxt][sOff + 0] = h0;
            *(uint4*)&sH[nxt][sOff + 8] = h1;
            *(uint4*)&sW[nxt][sOff + 0] = w0;
            *(uint4*)&sW[nxt][sOff + 8] = w1;
            __syncthreads();
        }
    }
#endif

    // --- epilogue: rank-1 input term + bias, activations, cell update
    const int j = colBase + l16;
    float wi[4], bb[4];
#pragma unroll
    for (int g = 0; g < 4; ++g) {
        wi[g] = Wih[g * HID + j];
        bb[g] = bias[g * HID + j];
    }

#pragma unroll
    for (int rf = 0; rf < 2; ++rf) {
#pragma unroll
        for (int r = 0; r < 8; ++r) {
            const int b = rowBase + waveM * 32 + rf * 16 + half * 8 + r;  // M = r + 8*(lane/16)
            const float xv = xbase[(size_t)b * xstride];
            const float gi = sigm(acc[0][rf][r] + xv * wi[0] + bb[0]);
            const float gf = sigm(acc[1][rf][r] + xv * wi[1] + bb[1]);
            const float gg = tanhf(acc[2][rf][r] + xv * wi[2] + bb[2]);
            const float go = sigm(acc[3][rf][r] + xv * wi[3] + bb[3]);
            const size_t idx = (size_t)b * HID + j;
            const float cn = gf * cstate[idx] + gi * gg;
            cstate[idx] = cn;
            hnext[idx]  = f2bf(go * tanhf(cn));
        }
    }
}

// ---------------------------------------------------------------------------
// Decoders: y[b] = dot(src[b,:], W_dec) + b_dec   (one wave per row)
// ---------------------------------------------------------------------------
__global__ __launch_bounds__(256)
void decode_bf16(const unsigned short* __restrict__ h, const float* __restrict__ wd,
                 const float* __restrict__ bd, float* __restrict__ y, int ystride)
{
    const int lane = threadIdx.x & 31;
    const int b = blockIdx.x * 8 + (threadIdx.x >> 5);
    const unsigned short* hp = h + (size_t)b * HID;
    float s = 0.f;
    for (int j = lane * 8; j < HID; j += 256) {
        uint4 q = *(const uint4*)(hp + j);
        const unsigned short* e = (const unsigned short*)&q;
        float4 w0 = *(const float4*)(wd + j);
        float4 w1 = *(const float4*)(wd + j + 4);
        s += bf2f(e[0]) * w0.x + bf2f(e[1]) * w0.y + bf2f(e[2]) * w0.z + bf2f(e[3]) * w0.w;
        s += bf2f(e[4]) * w1.x + bf2f(e[5]) * w1.y + bf2f(e[6]) * w1.z + bf2f(e[7]) * w1.w;
    }
    for (int m = 16; m; m >>= 1) s += __shfl_xor(s, m, 32);
    if (lane == 0) y[(size_t)b * ystride] = s + bd[0];
}

__global__ __launch_bounds__(256)
void decode_f32(const float* __restrict__ c, const float* __restrict__ wd,
                const float* __restrict__ bd, float* __restrict__ y, int ystride)
{
    const int lane = threadIdx.x & 31;
    const int b = blockIdx.x * 8 + (threadIdx.x >> 5);
    const float* cp = c + (size_t)b * HID;
    float s = 0.f;
    for (int j = lane * 4; j < HID; j += 128) {
        float4 cv = *(const float4*)(cp + j);
        float4 wv = *(const float4*)(wd + j);
        s += cv.x * wv.x + cv.y * wv.y + cv.z * wv.z + cv.w * wv.w;
    }
    for (int m = 16; m; m >>= 1) s += __shfl_xor(s, m, 32);
    if (lane == 0) y[(size_t)b * ystride] = s + bd[0];
}

// ---------------------------------------------------------------------------
// One-time prep: f32 -> bf16 weights, combined biases, zero h0/c0.
// ---------------------------------------------------------------------------
__global__ void prep(const float* __restrict__ Wp, const float* __restrict__ Wf,
                     const float* __restrict__ bihp, const float* __restrict__ bhhp,
                     const float* __restrict__ bihf, const float* __restrict__ bhhf,
                     unsigned short* __restrict__ Wpb, unsigned short* __restrict__ Wfb,
                     float* __restrict__ bp, float* __restrict__ bfc,
                     unsigned short* __restrict__ h0, float* __restrict__ c0)
{
    const size_t i0 = (size_t)blockIdx.x * blockDim.x + threadIdx.x;
    const size_t stride = (size_t)gridDim.x * blockDim.x;
    for (size_t k = i0; k < (size_t)4 * HID * HID; k += stride) {
        Wpb[k] = f2bf(Wp[k]);
        Wfb[k] = f2bf(Wf[k]);
    }
    for (size_t k = i0; k < (size_t)4 * HID; k += stride) {
        bp[k]  = bihp[k] + bhhp[k];
        bfc[k] = bihf[k] + bhhf[k];
    }
    for (size_t k = i0; k < (size_t)BATCH * HID; k += stride) {
        h0[k] = 0;        // bf16 zero
        c0[k] = 0.0f;
    }
}

// ---------------------------------------------------------------------------
extern "C" void kernel_launch(void* const* d_in, const int* in_sizes, int n_in,
                              void* d_out, int out_size, void* d_ws, size_t ws_size,
                              hipStream_t stream)
{
    const float* input_seq = (const float*)d_in[0];   // [B, 96, 1]
    const float* Wihp      = (const float*)d_in[1];   // [4H, 1]
    const float* Whhp      = (const float*)d_in[2];   // [4H, H]
    const float* bihp      = (const float*)d_in[3];
    const float* bhhp      = (const float*)d_in[4];
    const float* Wihf      = (const float*)d_in[5];
    const float* Whhf      = (const float*)d_in[6];
    const float* bihf      = (const float*)d_in[7];
    const float* bhhf      = (const float*)d_in[8];
    const float* Wdec      = (const float*)d_in[9];   // [1, H]
    const float* bdec      = (const float*)d_in[10];  // [1]
    float* out = (float*)d_out;                       // [B, 128, 1]

    // Carve workspace (~25.2 MB)
    char* ws = (char*)d_ws;
    size_t o = 0;
    auto carve = [&](size_t bytes) -> char* {
        char* p = ws + o;
        o = (o + bytes + 255) & ~(size_t)255;
        return p;
    };
    unsigned short* Wpb = (unsigned short*)carve((size_t)4 * HID * HID * 2);
    unsigned short* Wfb = (unsigned short*)carve((size_t)4 * HID * HID * 2);
    float* bp   = (float*)carve((size_t)4 * HID * 4);
    float* bfc  = (float*)carve((size_t)4 * HID * 4);
    unsigned short* hb0 = (unsigned short*)carve((size_t)BATCH * HID * 2);
    unsigned short* hb1 = (unsigned short*)carve((size_t)BATCH * HID * 2);
    float* cst  = (float*)carve((size_t)BATCH * HID * 4);
    float* xbuf = (float*)carve((size_t)BATCH * 4);

    prep<<<4096, 256, 0, stream>>>(Whhp, Whhf, bihp, bhhp, bihf, bhhf,
                                   Wpb, Wfb, bp, bfc, hb0, cst);

    const dim3 sg(HID / 32, BATCH / 128);   // 32 x 8 = 256 blocks

    // Past phase: x_t from input_seq (element stride 96)
    for (int t = 0; t < TPAST; ++t) {
        const unsigned short* hin = (t & 1) ? hb1 : hb0;
        unsigned short*      hout = (t & 1) ? hb0 : hb1;
        lstm_step<<<sg, 256, 0, stream>>>(hin, cst, hout, Wpb, Wihp, bp,
                                          input_seq + t, TPAST);
        decode_bf16<<<BATCH / 8, 256, 0, stream>>>(hout, Wdec, bdec, out + t, TTOT);
    }
    // Future phase: x_t = decode(c) (faithful to reference), then cell update
    for (int s = TPAST; s < TTOT; ++s) {
        const unsigned short* hin = (s & 1) ? hb1 : hb0;
        unsigned short*      hout = (s & 1) ? hb0 : hb1;
        decode_f32<<<BATCH / 8, 256, 0, stream>>>(cst, Wdec, bdec, xbuf, 1);
        lstm_step<<<sg, 256, 0, stream>>>(hin, cst, hout, Wfb, Wihf, bfc, xbuf, 1);
        decode_bf16<<<BATCH / 8, 256, 0, stream>>>(hout, Wdec, bdec, out + s, TTOT);
    }
}
